// Zig_MambaBlock_11252814315695
// MI455X (gfx1250) — compile-verified, matrix operand order checked by reading, not run on hardware
//
#include <hip/hip_runtime.h>
#include <hip/hip_bf16.h>
#include <math.h>

// ---- problem constants (match reference) ----
#define B_SZ 4
#define SEQ  1024
#define DIM  1024
#define DI   2048
#define DS   16
#define DTR  64
#define DC   4
#define XPN  (DTR + 2 * DS)   // 96
#define BSR  (B_SZ * SEQ)     // 4096 rows
#define EPSV 1e-5f

typedef _Float16 h16v16 __attribute__((ext_vector_type(16)));
typedef _Float16 h16v8  __attribute__((ext_vector_type(8)));
typedef _Float16 h16v2  __attribute__((ext_vector_type(2)));
typedef float    f32v8  __attribute__((ext_vector_type(8)));
typedef unsigned int u32x4 __attribute__((ext_vector_type(4)));
typedef int          i32x4 __attribute__((ext_vector_type(4)));
typedef int          i32x8 __attribute__((ext_vector_type(8)));

// Tensor Data Mover availability (device pass only). ROCm 7.2 declares the
// 5-arg builtin; the amdgpu-toolchain (therock headers) declares 6 args and
// is the only lane that ships the TDM header -> use that to discriminate.
#if defined(__gfx1250__) && __has_builtin(__builtin_amdgcn_tensor_load_to_lds)
#define USE_TDM 1
#if __has_include(<hip/amd_detail/amd_gfx1250_TDM.h>)
#define TDM_6ARG 1
#endif
#endif

__device__ __forceinline__ float silu_f(float v) {
    return v / (1.f + __expf(-v));
}
__device__ __forceinline__ float softplus_f(float v) {
    return (v > 20.f) ? v : log1pf(__expf(v));
}

// ---------------------------------------------------------------------------
// f32 -> f16 weight conversion
// ---------------------------------------------------------------------------
__global__ void f32_to_f16_kernel(const float* __restrict__ src,
                                  _Float16* __restrict__ dst, int n) {
    int i = blockIdx.x * blockDim.x + threadIdx.x;
    if (i < n) dst[i] = (_Float16)src[i];
}

// ---------------------------------------------------------------------------
// mod = silu(c) @ W_ada + b_ada   (tiny: 4 x 3072, K = 2048)
// ---------------------------------------------------------------------------
__global__ void adaln_mod_kernel(const float* __restrict__ c,
                                 const float* __restrict__ W_ada,
                                 const float* __restrict__ b_ada,
                                 float* __restrict__ modv) {
    int idx = blockIdx.x * blockDim.x + threadIdx.x;   // B_SZ * 3*DIM
    if (idx >= B_SZ * 3 * DIM) return;
    int j = idx % (3 * DIM);
    int b = idx / (3 * DIM);
    float acc = b_ada[j];
    const float* crow = c + b * 2 * DIM;
    for (int k = 0; k < 2 * DIM; ++k) {
        acc += silu_f(crow[k]) * W_ada[(size_t)k * (3 * DIM) + j];
    }
    modv[idx] = acc;
}

// ---------------------------------------------------------------------------
// h[b, t, :] = LN(x[b, perm[t], :]) * (1 + scale[b]) + shift[b]   (f16 out)
// one block per (b, t)
// ---------------------------------------------------------------------------
__global__ void __launch_bounds__(256)
ln_mod_perm_kernel(const float* __restrict__ x,
                   const float* __restrict__ gamma,
                   const float* __restrict__ beta,
                   const float* __restrict__ modv,
                   const int* __restrict__ perm,
                   _Float16* __restrict__ h) {
    __shared__ float red[256];
    __shared__ float red2[256];
    const int bid = blockIdx.x;
    const int b = bid >> 10;
    const int t = bid & 1023;
    const int s = perm[t];
    const float* xr = x + ((size_t)(b << 10) + s) * DIM;
    float sum = 0.f, sq = 0.f;
    for (int i = threadIdx.x; i < DIM; i += 256) {
        float v = xr[i];
        sum += v;
        sq  += v * v;
    }
    red[threadIdx.x] = sum;
    red2[threadIdx.x] = sq;
    __syncthreads();
    for (int o = 128; o > 0; o >>= 1) {
        if ((int)threadIdx.x < o) {
            red[threadIdx.x]  += red[threadIdx.x + o];
            red2[threadIdx.x] += red2[threadIdx.x + o];
        }
        __syncthreads();
    }
    const float mean = red[0] * (1.f / DIM);
    const float var  = red2[0] * (1.f / DIM) - mean * mean;
    const float rstd = rsqrtf(var + EPSV);
    const float* modb = modv + b * 3 * DIM;   // [shift | scale | gate]
    _Float16* hr = h + ((size_t)(b << 10) + t) * DIM;
    for (int i = threadIdx.x; i < DIM; i += 256) {
        float v = (xr[i] - mean) * rstd * gamma[i] + beta[i];
        v = v * (1.f + modb[DIM + i]) + modb[i];
        hr[i] = (_Float16)v;
    }
}

// ---------------------------------------------------------------------------
// WMMA GEMM: C[M,N] = A[M,K](f16, row stride lda) x B[K,N](f16, row-major)
// 128 threads = 4 waves; 64x64 tile; K step 32; v_wmma_f32_16x16x32_f16.
// A tile staged via Tensor Data Mover (TDM) when available, else VALU loads.
// Epilogue modes:
//   0: store f32            1: store f16
//   2: store f32 softplus(acc + bias[n])
//   3: finalOut[b, perm[t], n] = xres[...] + gate[b,n] * acc  (gate = modv[:,2D:])
// ---------------------------------------------------------------------------
#define TM 64
#define TN 64
#define TK 32
#define LDP 40   // padded LDS row stride for B (elements)

__global__ void __launch_bounds__(128)
wmma_gemm_kernel(const _Float16* __restrict__ A, const _Float16* __restrict__ B,
                 int M, int N, int K, int lda, int mode,
                 float* __restrict__ outF32, _Float16* __restrict__ outF16,
                 const float* __restrict__ bias,
                 const float* __restrict__ xres, const float* __restrict__ modv,
                 const int* __restrict__ perm, float* __restrict__ finalOut) {
    __shared__ _Float16 Alds[TM][TK];    // dense: TDM writes the tile directly
    __shared__ _Float16 Blds[TN][LDP];   // transposed: [n][k]

    const int tid  = threadIdx.x;
    const int wave = tid >> 5;
    const int lane = tid & 31;
    const int gm0  = blockIdx.y * TM;
    const int gn0  = blockIdx.x * TN;
    const int kb   = (lane >= 16) ? 8 : 0;
    const int nl   = lane & 15;

    f32v8 acc[4] = {};

    for (int k0 = 0; k0 < K; k0 += TK) {
        // ---- stage A tile 64x32 ----
#ifdef USE_TDM
        if (wave == 0) {
            // Build D# for a 2-D tile: 64 rows x 32 f16, row stride = lda.
            const unsigned long long gaddr =
                (unsigned long long)(const void*)(A + (size_t)gm0 * lda + k0);
            const unsigned int ldsaddr =
                (unsigned int)(unsigned long long)(void*)&Alds[0][0];
            u32x4 g0;
            g0[0] = 1u;                                   // count = 1 (valid D#)
            g0[1] = ldsaddr;                              // lds_addr (bytes)
            g0[2] = (unsigned int)(gaddr & 0xFFFFFFFFu);  // global_addr[31:0]
            g0[3] = (unsigned int)((gaddr >> 32) & 0x01FFFFFFu) | (2u << 30); // [56:32] | type=2
            i32x8 g1;
            g1[0] = (int)(1u << 16);              // data_size = 1 -> 2 bytes/elem
            g1[1] = (int)((TK & 0xFFFF) << 16);   // tensor_dim0[15:0]
            g1[2] = (int)((TM & 0xFFFF) << 16);   // tensor_dim0[31:16]=0 | tensor_dim1[15:0]
            g1[3] = (int)((TK & 0xFFFF) << 16);   // tensor_dim1[31:16]=0 | tile_dim0
            g1[4] = (int)(TM & 0xFFFF);           // tile_dim1 | tile_dim2 = 0
            g1[5] = lda;                          // tensor_dim0_stride[31:0]
            g1[6] = 0;                            // stride0[47:32] | stride1[15:0]
            g1[7] = 0;
            i32x4 z4 = {0, 0, 0, 0};
#ifdef TDM_6ARG
            i32x8 z8 = {0, 0, 0, 0, 0, 0, 0, 0};
            __builtin_amdgcn_tensor_load_to_lds(g0, g1, z4, z4, z8, 0);
#else
            __builtin_amdgcn_tensor_load_to_lds(g0, g1, z4, z4, 0);
#endif
            __builtin_amdgcn_s_wait_tensorcnt(0);
        }
#else
        {
            const int row = tid >> 1;
            const int col = (tid & 1) * 16;
            const _Float16* src = A + (size_t)(gm0 + row) * lda + k0 + col;
            *(h16v8*)&Alds[row][col]     = *(const h16v8*)(src);
            *(h16v8*)&Alds[row][col + 8] = *(const h16v8*)(src + 8);
        }
#endif
        // ---- stage B tile 32x64, transposed into [n][k] ----
        for (int cidx = tid; cidx < 256; cidx += 128) {
            const int krow = cidx >> 3;
            const int coff = (cidx & 7) * 8;
            h16v8 v;
            if (gn0 + coff < N) {
                const _Float16* bsrc = B + (size_t)(k0 + krow) * N + gn0 + coff;
                v = *(const h16v8*)(bsrc);
                if (k0 + TK < K)
                    __builtin_prefetch(bsrc + (size_t)TK * N, 0, 1); // global_prefetch_b8
            } else {
#pragma unroll
                for (int j = 0; j < 8; ++j) v[j] = (_Float16)0.f;
            }
#pragma unroll
            for (int j = 0; j < 8; ++j) Blds[coff + j][krow] = v[j];
        }
        __syncthreads();

        // ---- A fragment (ISA 16-bit A 16x32 layout) ----
        h16v16 afrag;
        {
            const _Float16* ar = &Alds[wave * 16 + nl][0];
#pragma unroll
            for (int v = 0; v < 8; ++v) {
                const int kk = (v < 4) ? (kb + 2 * v) : (16 + kb + 2 * (v - 4));
                h16v2 p = *(const h16v2*)(ar + kk);
                afrag[2 * v]     = p[0];
                afrag[2 * v + 1] = p[1];
            }
        }
        // ---- 4 N-tiles of 16 ----
#pragma unroll
        for (int nt = 0; nt < 4; ++nt) {
            h16v16 bfrag;
            const _Float16* br = &Blds[nt * 16 + nl][0];
#pragma unroll
            for (int v = 0; v < 8; ++v) {
                const int kk = (v < 4) ? (kb + 2 * v) : (16 + kb + 2 * (v - 4));
                h16v2 p = *(const h16v2*)(br + kk);
                bfrag[2 * v]     = p[0];
                bfrag[2 * v + 1] = p[1];
            }
            acc[nt] = __builtin_amdgcn_wmma_f32_16x16x32_f16(
                false, afrag, false, bfrag, (short)0, acc[nt], false, false);
        }
        __syncthreads();
    }

    // ---- epilogue: VGPR j -> M = j (lanes 0-15) / j+8 (lanes 16-31) ----
    const int mbase = wave * 16 + ((lane >= 16) ? 8 : 0);
#pragma unroll
    for (int nt = 0; nt < 4; ++nt) {
        const int gn = gn0 + nt * 16 + nl;
        if (gn >= N) continue;
#pragma unroll
        for (int j = 0; j < 8; ++j) {
            const int gm = gm0 + mbase + j;
            const float v = acc[nt][j];
            if (mode == 0) {
                outF32[(size_t)gm * N + gn] = v;
            } else if (mode == 1) {
                outF16[(size_t)gm * N + gn] = (_Float16)v;
            } else if (mode == 2) {
                outF32[(size_t)gm * N + gn] = softplus_f(v + bias[gn]);
            } else {
                const int b = gm >> 10;
                const int t = gm & 1023;
                const int s = perm[t];
                const size_t o = ((size_t)(b << 10) + s) * (size_t)N + gn;
                const float g = modv[b * 3 * DIM + 2 * DIM + gn];
                finalOut[o] = xres[o] + g * v;
            }
        }
    }
}

// ---------------------------------------------------------------------------
// causal depthwise conv (DC=4) along permuted sequence + bias + SiLU -> f16
// xz rows are (2*DI); xm = first DI columns
// ---------------------------------------------------------------------------
__global__ void conv_silu_kernel(const float* __restrict__ xz,
                                 const float* __restrict__ conv_w,
                                 const float* __restrict__ conv_b,
                                 _Float16* __restrict__ xmc) {
    const int idx = blockIdx.x * blockDim.x + threadIdx.x;  // BSR*DI
    if (idx >= BSR * DI) return;
    const int ch = idx % DI;
    const int r  = idx / DI;
    const int t  = r % SEQ;
    const int b  = r / SEQ;
    float acc = conv_b[ch];
#pragma unroll
    for (int j = 0; j < DC; ++j) {
        const int tt = t - (DC - 1) + j;
        if (tt >= 0)
            acc += conv_w[ch * DC + j] *
                   xz[((size_t)(b * SEQ + tt)) * (2 * DI) + ch];
    }
    xmc[(size_t)r * DI + ch] = (_Float16)silu_f(acc);
}

// ---------------------------------------------------------------------------
// selective scan: one thread per (b, ch); 16 fp32 states in registers;
// B_t/C_t broadcast through LDS; sequential over SEQ.
// ---------------------------------------------------------------------------
__global__ void __launch_bounds__(256)
scan_kernel(const float* __restrict__ dt,
            const _Float16* __restrict__ xmc,
            const _Float16* __restrict__ xdbl,   // (BSR, 96) = [dtr | B | C]
            const float* __restrict__ A_log,
            float* __restrict__ yraw) {
    const int b = blockIdx.x >> 3;       // 8 chunks of 256 channels per batch
    const int chunk = blockIdx.x & 7;
    const int ch = chunk * 256 + threadIdx.x;

    float Arow[DS];
#pragma unroll
    for (int s = 0; s < DS; ++s) Arow[s] = -__expf(A_log[ch * DS + s]);
    float hs[DS];
#pragma unroll
    for (int s = 0; s < DS; ++s) hs[s] = 0.f;

    __shared__ float Bs[DS];
    __shared__ float Cs[DS];

    for (int t = 0; t < SEQ; ++t) {
        const size_t r = (size_t)(b * SEQ + t);
        if (threadIdx.x < DS)
            Bs[threadIdx.x] = (float)xdbl[r * XPN + DTR + threadIdx.x];
        else if (threadIdx.x < 2 * DS)
            Cs[threadIdx.x - DS] =
                (float)xdbl[r * XPN + DTR + DS + (threadIdx.x - DS)];
        __syncthreads();

        const float dtv = dt[r * DI + ch];
        const float dx  = dtv * (float)xmc[r * DI + ch];
        float y = 0.f;
#pragma unroll
        for (int s = 0; s < DS; ++s) {
            hs[s] = hs[s] * __expf(dtv * Arow[s]) + dx * Bs[s];
            y += hs[s] * Cs[s];
        }
        yraw[r * DI + ch] = y;
        __syncthreads();
    }
}

// ---------------------------------------------------------------------------
// yg = (yraw + xmc * D_param) * silu(z)  -> f16 ;  z = xz[:, DI:2*DI]
// ---------------------------------------------------------------------------
__global__ void combine_kernel(const float* __restrict__ yraw,
                               const _Float16* __restrict__ xmc,
                               const float* __restrict__ Dp,
                               const float* __restrict__ xz,
                               _Float16* __restrict__ yg) {
    const int idx = blockIdx.x * blockDim.x + threadIdx.x;  // BSR*DI
    if (idx >= BSR * DI) return;
    const int ch = idx % DI;
    const size_t r = (size_t)(idx / DI);
    const float z = xz[r * (2 * DI) + DI + ch];
    const float y = yraw[idx] + (float)xmc[idx] * Dp[ch];
    yg[idx] = (_Float16)(y * silu_f(z));
}

// ---------------------------------------------------------------------------
// host launcher
// ---------------------------------------------------------------------------
extern "C" void kernel_launch(void* const* d_in, const int* in_sizes, int n_in,
                              void* d_out, int out_size, void* d_ws, size_t ws_size,
                              hipStream_t stream) {
    (void)in_sizes; (void)n_in; (void)out_size; (void)ws_size;
    const float* x      = (const float*)d_in[0];
    const float* c      = (const float*)d_in[1];
    const float* W_ada  = (const float*)d_in[2];
    const float* b_ada  = (const float*)d_in[3];
    const float* gamma  = (const float*)d_in[4];
    const float* beta   = (const float*)d_in[5];
    const float* W_in   = (const float*)d_in[6];
    const float* conv_w = (const float*)d_in[7];
    const float* conv_b = (const float*)d_in[8];
    const float* W_xp   = (const float*)d_in[9];
    const float* W_dt   = (const float*)d_in[10];
    const float* b_dt   = (const float*)d_in[11];
    const float* A_log  = (const float*)d_in[12];
    const float* D_par  = (const float*)d_in[13];
    const float* W_out  = (const float*)d_in[14];
    const int*   perm   = (const int*)d_in[15];
    float* out = (float*)d_out;

    char* ws = (char*)d_ws;
    size_t cur = 0;
    auto alloc = [&](size_t bytes) -> void* {
        void* p = ws + cur;
        cur = (cur + bytes + 255) & ~(size_t)255;
        return p;
    };

    float*    modv   = (float*)   alloc((size_t)B_SZ * 3 * DIM * 4);
    _Float16* h16    = (_Float16*)alloc((size_t)BSR * DIM * 2);
    _Float16* Win16  = (_Float16*)alloc((size_t)DIM * 2 * DI * 2);
    _Float16* Wxp16  = (_Float16*)alloc((size_t)DI * XPN * 2);
    _Float16* Wdt16  = (_Float16*)alloc((size_t)DTR * DI * 2);
    _Float16* Wout16 = (_Float16*)alloc((size_t)DI * DIM * 2);
    float*    xz     = (float*)   alloc((size_t)BSR * 2 * DI * 4);
    _Float16* xmc    = (_Float16*)alloc((size_t)BSR * DI * 2);
    _Float16* xdbl   = (_Float16*)alloc((size_t)BSR * XPN * 2);
    float*    dt     = (float*)   alloc((size_t)BSR * DI * 4);
    float*    yraw   = (float*)   alloc((size_t)BSR * DI * 4);
    _Float16* yg     = (_Float16*)alloc((size_t)BSR * DI * 2);

    // weight conversions
    f32_to_f16_kernel<<<(DIM * 2 * DI + 255) / 256, 256, 0, stream>>>(W_in, Win16, DIM * 2 * DI);
    f32_to_f16_kernel<<<(DI * XPN + 255) / 256, 256, 0, stream>>>(W_xp, Wxp16, DI * XPN);
    f32_to_f16_kernel<<<(DTR * DI + 255) / 256, 256, 0, stream>>>(W_dt, Wdt16, DTR * DI);
    f32_to_f16_kernel<<<(DI * DIM + 255) / 256, 256, 0, stream>>>(W_out, Wout16, DI * DIM);

    // AdaLN modulation
    adaln_mod_kernel<<<(B_SZ * 3 * DIM + 255) / 256, 256, 0, stream>>>(c, W_ada, b_ada, modv);

    // LN + modulate + permute -> h (f16)
    ln_mod_perm_kernel<<<BSR, 256, 0, stream>>>(x, gamma, beta, modv, perm, h16);

    // xz = h @ W_in           (4096 x 4096, K=1024)
    wmma_gemm_kernel<<<dim3(2 * DI / TN, BSR / TM), 128, 0, stream>>>(
        h16, Win16, BSR, 2 * DI, DIM, DIM, 0,
        xz, nullptr, nullptr, nullptr, nullptr, nullptr, nullptr);

    // causal depthwise conv + SiLU -> xmc (f16)
    conv_silu_kernel<<<(BSR * DI + 255) / 256, 256, 0, stream>>>(xz, conv_w, conv_b, xmc);

    // xdbl = xmc @ W_xp       (4096 x 96, K=2048) -> f16
    wmma_gemm_kernel<<<dim3((XPN + TN - 1) / TN, BSR / TM), 128, 0, stream>>>(
        xmc, Wxp16, BSR, XPN, DI, DI, 1,
        nullptr, xdbl, nullptr, nullptr, nullptr, nullptr, nullptr);

    // dt = softplus(dtr @ W_dt + b_dt)   (4096 x 2048, K=64; A row stride 96)
    wmma_gemm_kernel<<<dim3(DI / TN, BSR / TM), 128, 0, stream>>>(
        xdbl, Wdt16, BSR, DI, DTR, XPN, 2,
        dt, nullptr, b_dt, nullptr, nullptr, nullptr, nullptr);

    // selective scan
    scan_kernel<<<B_SZ * (DI / 256), 256, 0, stream>>>(dt, xmc, xdbl, A_log, yraw);

    // yg = (yraw + xmc*D) * silu(z)
    combine_kernel<<<(BSR * DI + 255) / 256, 256, 0, stream>>>(yraw, xmc, D_par, xz, yg);

    // out[b, perm[t], :] = x[...] + gate * (yg @ W_out)   (4096 x 1024, K=2048)
    wmma_gemm_kernel<<<dim3(DIM / TN, BSR / TM), 128, 0, stream>>>(
        yg, Wout16, BSR, DIM, DI, DI, 3,
        nullptr, nullptr, nullptr, x, modv, perm, out);
}
